// ContrastiveLoss_26010321945303
// MI455X (gfx1250) — matrix operand order repaired and checked by reading, not compiled, not run on hardware
//
#include <hip/hip_runtime.h>

typedef __attribute__((ext_vector_type(16))) __bf16 v16bf;
typedef __attribute__((ext_vector_type(8)))  __bf16 v8bf;
typedef __attribute__((ext_vector_type(8)))  float  v8f;

#define NI 96    // images
#define NC 96    // recipes (captions)
#define NR 36    // regions per image
#define RP 48    // regions padded to 3x16
#define NW 48    // words per recipe
#define WP 64    // words padded to 2x32 (K granularity of bf16 WMMA)
#define ND 256   // feature dim
#define WAVES 4  // waves per block (one (c,i) pair per wave)

// ---------------------------------------------------------------------------
// Prep: fp32 -> bf16 staging buffers in workspace.
//   imgbf [96][48][256] : images, region rows 36..47 zeroed (GEMM1 B operand)
//   recbf [96][48][256] : recipes                          (GEMM1 A operand)
// ---------------------------------------------------------------------------
__global__ __launch_bounds__(256) void cvt_kernel(
    const float* __restrict__ img, const float* __restrict__ rec,
    __bf16* __restrict__ imgbf, __bf16* __restrict__ recbf) {
  int idx = blockIdx.x * 256 + threadIdx.x;
  const int NIMG = NI * RP * ND;
  if (idx < NIMG) {
    const int d = idx & (ND - 1);
    const int r = (idx >> 8) % RP;
    const int i = idx / (RP * ND);
    imgbf[idx] = (r < NR) ? (__bf16)img[(i * NR + r) * ND + d] : (__bf16)0.f;
  } else {
    idx -= NIMG;
    if (idx < NC * NW * ND) recbf[idx] = (__bf16)rec[idx];
  }
}

// ---------------------------------------------------------------------------
// Main: one wave computes score[i][c] for one (c,i) pair.
// ---------------------------------------------------------------------------
__global__ __launch_bounds__(128) void xattn_kernel(
    const float* __restrict__ images,     // [96][36][256] fp32 (cosine pass)
    const int*   __restrict__ cap_lens,   // [96]
    const __bf16* __restrict__ imgbf,     // [96][48][256] bf16 padded
    const __bf16* __restrict__ recbf,     // [96][48][256] bf16
    float* __restrict__ scores)           // [96][96], i-major
{
  __shared__ __bf16 sRecT[ND * WP];       // recipes[c] transposed [d][w], w 48..63 = 0  (32 KB)
  __shared__ __bf16 sAW[WAVES][RP * WP];  // per-wave attention weights (A of GEMM2)     (24 KB)

  const int c    = blockIdx.y;
  const int wave = threadIdx.x >> 5;
  const int lane = threadIdx.x & 31;
  const int ln   = lane & 15;             // position within lane-half
  const int kh   = lane >> 4;             // lane half (0/1)
  const int i    = blockIdx.x * WAVES + wave;

  // Build transposed recipe tile in LDS (GEMM2's B wants contiguous w per d).
  for (int idx = threadIdx.x; idx < WP * ND; idx += 128) {
    const int d = idx & (ND - 1);
    const int w = idx >> 8;
    sRecT[d * WP + w] = (w < NW) ? recbf[(c * NW + w) * ND + d] : (__bf16)0.f;
  }
  __syncthreads();

  const int cap = cap_lens[c];

  v8f zacc;
#pragma unroll
  for (int e = 0; e < 8; ++e) zacc[e] = 0.f;

  // =========================== GEMM1: attn = rec @ img^T ====================
  // A (16-bit 16x32 layout): lane m=ln holds row w, K = kh*8+0..7 (elem 0-7)
  //                          and K = 16+kh*8+0..7 (elem 8-15). From global recbf.
  // B (16-bit 32x16 layout): lane n=ln holds col r, K = kh*8+0..7 (elem 0-7)
  //                          and K = 16+kh*8+0..7 (elem 8-15). From global imgbf.
  v8f acc[3][3];
#pragma unroll
  for (int mt = 0; mt < 3; ++mt)
#pragma unroll
    for (int nt = 0; nt < 3; ++nt) acc[mt][nt] = zacc;

  const __bf16* gA = recbf + (size_t)c * NW * ND;
  const __bf16* gB = imgbf + (size_t)i * RP * ND;
  for (int k0 = 0; k0 < ND; k0 += 32) {
    v16bf A[3], B[3];
#pragma unroll
    for (int mt = 0; mt < 3; ++mt) {
      const int w = mt * 16 + ln;
      const __bf16* p = &gA[w * ND + k0 + kh * 8];
      v8bf lo = *(const v8bf*)p;
      v8bf hi = *(const v8bf*)(p + 16);
#pragma unroll
      for (int e = 0; e < 8; ++e) { A[mt][e] = lo[e]; A[mt][e + 8] = hi[e]; }
    }
#pragma unroll
    for (int nt = 0; nt < 3; ++nt) {
      const int r = nt * 16 + ln;
      const __bf16* q = &gB[r * ND + k0 + kh * 8];
      v8bf blo = *(const v8bf*)q;
      v8bf bhi = *(const v8bf*)(q + 16);
#pragma unroll
      for (int e = 0; e < 8; ++e) { B[nt][e] = blo[e]; B[nt][e + 8] = bhi[e]; }
    }
#pragma unroll
    for (int mt = 0; mt < 3; ++mt)
#pragma unroll
      for (int nt = 0; nt < 3; ++nt)
        acc[mt][nt] = __builtin_amdgcn_wmma_f32_16x16x32_bf16(
            false, A[mt], false, B[nt], (short)0, acc[mt][nt], false, false);
  }

  // ============ leaky-relu + mask + column L2 norm + softmax over w =========
  // D layout: VGPR j of tile (mt,nt): (w = mt*16 + kh*8 + j, r = nt*16 + ln).
  // A full r-column is split between lane ln and lane ln+16 -> shfl_xor(16).
#pragma unroll
  for (int nt = 0; nt < 3; ++nt) {
    float ss = 0.f;
#pragma unroll
    for (int mt = 0; mt < 3; ++mt)
#pragma unroll
      for (int j = 0; j < 8; ++j) {
        const int w = mt * 16 + kh * 8 + j;
        float v = acc[mt][nt][j];
        v = (v > 0.f) ? v : 0.1f * v;        // leaky_relu 0.1
        v = (w < cap) ? v : 0.f;             // word mask
        acc[mt][nt][j] = v;
        ss = fmaf(v, v, ss);
      }
    ss += __shfl_xor(ss, 16, 32);
    const float scale = 9.0f * __frsqrt_rn(fmaxf(ss, 1e-24f));  // lambda=9, /max(nrm,1e-12)

    float mx = -1e30f;
#pragma unroll
    for (int mt = 0; mt < 3; ++mt)
#pragma unroll
      for (int j = 0; j < 8; ++j) {
        const int w = mt * 16 + kh * 8 + j;
        if (w < cap) mx = fmaxf(mx, acc[mt][nt][j] * scale);
      }
    mx = fmaxf(mx, __shfl_xor(mx, 16, 32));

    float es = 0.f;
#pragma unroll
    for (int mt = 0; mt < 3; ++mt)
#pragma unroll
      for (int j = 0; j < 8; ++j) {
        const int w = mt * 16 + kh * 8 + j;
        const float e = (w < cap) ? __expf(acc[mt][nt][j] * scale - mx) : 0.f;
        acc[mt][nt][j] = e;
        es += e;
      }
    es += __shfl_xor(es, 16, 32);
    const float inv = 1.f / es;

    // Store aw^T row-major [r][w] bf16 (== GEMM2 A layout), pad w 48..63 = 0
    const int r = nt * 16 + ln;
    __bf16* row = &sAW[wave][r * WP];
#pragma unroll
    for (int mt = 0; mt < 3; ++mt)
#pragma unroll
      for (int j = 0; j < 8; ++j)
        row[mt * 16 + kh * 8 + j] = (__bf16)(acc[mt][nt][j] * inv);
#pragma unroll
    for (int j = 0; j < 8; ++j) row[NW + kh * 8 + j] = (__bf16)0.f;
  }

  // ========= GEMM2 (streamed over d): wctx = aw^T @ rec, fold reductions ====
  float numa[3][8], n2a[3][8], n1a[3][8];
#pragma unroll
  for (int mt = 0; mt < 3; ++mt)
#pragma unroll
    for (int j = 0; j < 8; ++j) { numa[mt][j] = 0.f; n2a[mt][j] = 0.f; n1a[mt][j] = 0.f; }

  const float* gI = images + (size_t)i * NR * ND;
  for (int dt = 0; dt < ND / 16; ++dt) {
    const int dbase = dt * 16;
    v8f c2[3];
    c2[0] = zacc; c2[1] = zacc; c2[2] = zacc;
#pragma unroll
    for (int kc = 0; kc < 2; ++kc) {          // K = w, padded 48 -> 64
      const int k0 = kc * 32;
      v16bf A2[3], B2;
#pragma unroll
      for (int mt = 0; mt < 3; ++mt) {
        const int r = mt * 16 + ln;
        const __bf16* p = &sAW[wave][r * WP + k0 + kh * 8];
        v8bf lo = *(const v8bf*)p;
        v8bf hi = *(const v8bf*)(p + 16);
#pragma unroll
        for (int e = 0; e < 8; ++e) { A2[mt][e] = lo[e]; A2[mt][e + 8] = hi[e]; }
      }
      // B[k=w][n=d]: lane holds d = dbase+ln, w = k0+kh*8+0..7 (elem 0-7) and k0+16+kh*8+0..7 (elem 8-15)
      {
        const __bf16* q = &sRecT[(dbase + ln) * WP + k0 + kh * 8];
        v8bf blo = *(const v8bf*)q;
        v8bf bhi = *(const v8bf*)(q + 16);
#pragma unroll
        for (int e = 0; e < 8; ++e) { B2[e] = blo[e]; B2[e + 8] = bhi[e]; }
      }
#pragma unroll
      for (int mt = 0; mt < 3; ++mt)
        c2[mt] = __builtin_amdgcn_wmma_f32_16x16x32_bf16(
            false, A2[mt], false, B2, (short)0, c2[mt], false, false);
    }
    // D layout: VGPR j of tile mt: (r = mt*16 + kh*8 + j, d = dbase + ln)
#pragma unroll
    for (int mt = 0; mt < 3; ++mt)
#pragma unroll
      for (int j = 0; j < 8; ++j) {
        const int r = mt * 16 + kh * 8 + j;
        const float iv = (r < NR) ? gI[r * ND + dbase + ln] : 0.f;
        const float wv = c2[mt][j];
        numa[mt][j] = fmaf(wv, iv, numa[mt][j]);
        n2a[mt][j]  = fmaf(wv, wv, n2a[mt][j]);
        n1a[mt][j]  = fmaf(iv, iv, n1a[mt][j]);
      }
  }

  // ======== reduce over d lanes, cosine per region, LSE over regions ========
  float ts[3][8];
#pragma unroll
  for (int mt = 0; mt < 3; ++mt)
#pragma unroll
    for (int j = 0; j < 8; ++j) {
      float a = numa[mt][j], b = n2a[mt][j], n1 = n1a[mt][j];
#pragma unroll
      for (int off = 1; off < 16; off <<= 1) {
        a  += __shfl_xor(a, off, 32);
        b  += __shfl_xor(b, off, 32);
        n1 += __shfl_xor(n1, off, 32);
      }
      const float den = fmaxf(sqrtf(n1) * sqrtf(b), 1e-8f);
      ts[mt][j] = 6.0f * (a / den);          // LAMBDA_LSE * row_sim
    }
  float mx = -1e30f;
#pragma unroll
  for (int mt = 0; mt < 3; ++mt)
#pragma unroll
    for (int j = 0; j < 8; ++j) {
      const int r = mt * 16 + kh * 8 + j;
      if (r < NR) mx = fmaxf(mx, ts[mt][j]);
    }
  mx = fmaxf(mx, __shfl_xor(mx, 16, 32));
  float s = 0.f;
#pragma unroll
  for (int mt = 0; mt < 3; ++mt)
#pragma unroll
    for (int j = 0; j < 8; ++j) {
      const int r = mt * 16 + kh * 8 + j;
      if (r < NR) s += __expf(ts[mt][j] - mx);
    }
  s += __shfl_xor(s, 16, 32);
  const float sim = (mx + __logf(s)) * (1.0f / 6.0f);
  if (lane == 0) scores[i * NC + c] = sim;   // scores = sim.T
}

// ---------------------------------------------------------------------------
// Loss: triplet margin reduction of the 96x96 score matrix -> scalar.
// ---------------------------------------------------------------------------
__global__ __launch_bounds__(256) void loss_kernel(
    const float* __restrict__ scores, float* __restrict__ out) {
  __shared__ float diag[NC];
  __shared__ float red[256];
  const int t = threadIdx.x;
  if (t < NC) diag[t] = scores[t * NC + t];
  __syncthreads();
  float acc = 0.f;
  for (int idx = t; idx < NI * NC; idx += 256) {
    const int ii = idx / NC, cc = idx % NC;
    if (ii != cc) {
      const float sv = scores[idx];
      acc += fmaxf(0.2f + sv - diag[cc], 0.f);   // cost_image  (diag[None,:])
      acc += fmaxf(0.2f + sv - diag[ii], 0.f);   // cost_recipe (diag[:,None])
    }
  }
  red[t] = acc;
  __syncthreads();
  for (int o = 128; o > 0; o >>= 1) {
    if (t < o) red[t] += red[t + o];
    __syncthreads();
  }
  if (t == 0) out[0] = red[0];
}

// ---------------------------------------------------------------------------
extern "C" void kernel_launch(void* const* d_in, const int* in_sizes, int n_in,
                              void* d_out, int out_size, void* d_ws, size_t ws_size,
                              hipStream_t stream) {
  const float* images  = (const float*)d_in[0];   // (96,36,256) fp32
  const float* recipes = (const float*)d_in[1];   // (96,48,256) fp32
  const int*   cap     = (const int*)d_in[2];     // (96,) int32

  __bf16* imgbf  = (__bf16*)d_ws;                                        // 2.25 MB
  __bf16* recbf  = imgbf + (size_t)NI * RP * ND;                         // 2.25 MB
  float*  scores = (float*)(recbf + (size_t)NC * NW * ND);               // 36 KB

  const int ncv = NI * RP * ND + NC * NW * ND;
  cvt_kernel<<<(ncv + 255) / 256, 256, 0, stream>>>(images, recipes, imgbf, recbf);

  dim3 grid(NI / WAVES, NC);   // (24, 96), 128 threads = 4 wave32 per block
  xattn_kernel<<<grid, 128, 0, stream>>>(images, cap, imgbf, recbf, scores);

  loss_kernel<<<1, 256, 0, stream>>>(scores, (float*)d_out);
}
